// HybridGNNModel_20830591386075
// MI455X (gfx1250) — matrix operand order repaired and checked by reading, not compiled, not run on hardware
//
#include <hip/hip_runtime.h>
#include <hip/hip_bf16.h>

// ---------------------------------------------------------------------------
// HybridGNN pipeline for gfx1250 (MI455X, wave32, WMMA).
// Dense GEMMs: v_wmma_f32_16x16x32_bf16 (fp32 accumulate), one wave per
// 16x64 output strip (4 WMMAs per k-step sharing one A fragment).
// Weights are pre-packed to bf16 in WMMA B-fragment order so each B fragment
// is a single aligned 32B vector load.  Edge softmax/aggregation: 3-pass
// atomic segment softmax with wave32-per-edge coalesced scatter.
// ---------------------------------------------------------------------------

typedef __attribute__((ext_vector_type(16))) __bf16 v16bf;
typedef __attribute__((ext_vector_type(8)))  float  v8f;

static constexpr int Nn    = 50000;
static constexpr int Ee    = 400000;
static constexpr int DIN   = 128;
static constexpr int H0    = 128;
static constexpr int HEADS = 4;
static constexpr int C1    = 64;
static constexpr int C2    = 32;
static constexpr int C3    = 64;
static constexpr int DOUT  = 32;

__device__ __forceinline__ float waveReduceSum(float v) {
  // wave32: full reduction in 5 shuffles
  #pragma unroll
  for (int o = 16; o > 0; o >>= 1) v += __shfl_down(v, o, 32);
  return v;
}

__device__ __forceinline__ void atomicMaxF(float* addr, float val) {
  // IEEE-754 monotonic bit trick: int max for >=0, unsigned min for <0.
  if (val >= 0.0f) {
    atomicMax((int*)addr, __float_as_int(val));
  } else {
    atomicMin((unsigned int*)addr, (unsigned int)__float_as_int(val));
  }
}

// ---------------------------------------------------------------------------
// Pack W[K x Nout] (fp32, row-major) into bf16 WMMA B-fragment order:
// element index ((colTile*Ksteps + kstep)*32 + lane) holds the v16bf that
// `lane` feeds to v_wmma for that (colTile, kstep).  One thread per fragment.
// B 32x16 bf16 layout: lanes 0-15 K={0..7,16..23} (2/VGPR), lanes 16-31 +8.
// ---------------------------------------------------------------------------
__global__ void pack_b_kernel(const float* __restrict__ W, __bf16* __restrict__ P,
                              int K, int Nout) {
  int idx = blockIdx.x * blockDim.x + threadIdx.x;
  int ksteps = K >> 5;
  int total = (Nout >> 4) * ksteps * 32;
  if (idx >= total) return;
  int lane = idx & 31;
  int rest = idx >> 5;
  int ks = rest % ksteps;
  int ct = rest / ksteps;
  int half = lane >> 4, r = lane & 15;
  int col = ct * 16 + r;
  __bf16* out = P + (size_t)idx * 16;
  #pragma unroll
  for (int v = 0; v < 8; ++v) {
    int kb = ks * 32 + 8 * half + 2 * v + (v >= 4 ? 8 : 0);
    out[2 * v]     = (__bf16)W[(size_t)kb       * Nout + col];
    out[2 * v + 1] = (__bf16)W[(size_t)(kb + 1) * Nout + col];
  }
}

// ---------------------------------------------------------------------------
// GEMM: C[M x Nout] = A[M x K] @ W[K x Nout] (+bias)
// bf16 WMMA, fp32 accumulate.  One wave computes a 16 x (CT*16) output strip:
// A fragment built from 4x float4 loads per 32-wide k-step, reused by CT
// independent WMMAs; each B fragment is one aligned v16bf (32B) load from
// the packed weight buffer.  M%16==0, K%32==0, Nout%(CT*16)==0.
// ---------------------------------------------------------------------------
template <int CT, bool BIAS>
__global__ void gemm_wmma_kernel(const float* __restrict__ A,
                                 const __bf16* __restrict__ Bp,
                                 const float* __restrict__ bias,
                                 float* __restrict__ C,
                                 int M, int K, int Nout) {
  const int lane = threadIdx.x & 31;
  const int wave = threadIdx.x >> 5;
  const int stripCols  = CT * 16;
  const int colStrips  = Nout / stripCols;
  const int nStrips    = (M >> 4) * colStrips;
  int strip = blockIdx.x * (blockDim.x >> 5) + wave;
  if (strip >= nStrips) return;
  const int rowT = strip / colStrips;
  const int colS = strip - rowT * colStrips;
  const int half = lane >> 4;     // 0: lanes 0-15, 1: lanes 16-31
  const int r    = lane & 15;
  const int ksteps = K >> 5;
  const float* Arow = A + (size_t)(rowT * 16 + r) * K;   // A-fragment row
  const v16bf* Bpack = (const v16bf*)Bp;

  v8f acc[CT];
  #pragma unroll
  for (int t = 0; t < CT; ++t) acc[t] = (v8f){};

  for (int ks = 0; ks < ksteps; ++ks) {
    const int k0 = ks * 32;
    // A 16x32 bf16 fragment: lanes 0-15 hold K {0..7,16..23}, lanes 16-31 +8.
    // Two contiguous 8-float chunks -> 4 x 16B-aligned float4 loads.
    const float* a01 = Arow + k0 + 8 * half;
    const float* a23 = a01 + 16;
    float4 f0 = *(const float4*)(a01);
    float4 f1 = *(const float4*)(a01 + 4);
    float4 f2 = *(const float4*)(a23);
    float4 f3 = *(const float4*)(a23 + 4);
    v16bf a;
    a[0]  = (__bf16)f0.x;  a[1]  = (__bf16)f0.y;
    a[2]  = (__bf16)f0.z;  a[3]  = (__bf16)f0.w;
    a[4]  = (__bf16)f1.x;  a[5]  = (__bf16)f1.y;
    a[6]  = (__bf16)f1.z;  a[7]  = (__bf16)f1.w;
    a[8]  = (__bf16)f2.x;  a[9]  = (__bf16)f2.y;
    a[10] = (__bf16)f2.z;  a[11] = (__bf16)f2.w;
    a[12] = (__bf16)f3.x;  a[13] = (__bf16)f3.y;
    a[14] = (__bf16)f3.z;  a[15] = (__bf16)f3.w;
    // CT independent WMMAs sharing the A fragment; B = one 32B vector load.
    #pragma unroll
    for (int t = 0; t < CT; ++t) {
      v16bf b = Bpack[((size_t)(colS * CT + t) * ksteps + ks) * 32 + lane];
      acc[t] = __builtin_amdgcn_wmma_f32_16x16x32_bf16(
          /*neg_a=*/false, a, /*neg_b=*/false, b,
          /*c_mod=*/(short)0, acc[t], /*reuse_a=*/false, /*reuse_b=*/false);
    }
  }

  #pragma unroll
  for (int t = 0; t < CT; ++t) {
    int ncol = colS * stripCols + t * 16 + r;
    float bv = BIAS ? bias[ncol] : 0.0f;
    #pragma unroll
    for (int v = 0; v < 8; ++v) {
      // D f32 layout: M = v + 8*half, N = lane&15
      int mrow = rowT * 16 + v + 8 * half;
      C[(size_t)mrow * Nout + ncol] = acc[t][v] + bv;
    }
  }
}

// s[n] = dot(h[n,:128], att)  — wave per node
__global__ void matvec128_kernel(const float* __restrict__ h,
                                 const float* __restrict__ att,
                                 float* __restrict__ s, int N) {
  int node = (int)((blockIdx.x * (size_t)blockDim.x + threadIdx.x) >> 5);
  int lane = threadIdx.x & 31;
  if (node >= N) return;
  const float* row = h + (size_t)node * 128;
  float p = row[lane] * att[lane] + row[lane + 32] * att[lane + 32]
          + row[lane + 64] * att[lane + 64] + row[lane + 96] * att[lane + 96];
  p = waveReduceSum(p);
  if (lane == 0) s[node] = p;
}

// GAT per-node attention scores: a_s[n,h]=dot(h[n,h,:],att_src[h]), same a_d.
__global__ void gat_node_scores_kernel(const float* __restrict__ h,
                                       const float* __restrict__ att_src,
                                       const float* __restrict__ att_dst,
                                       float* __restrict__ a_s,
                                       float* __restrict__ a_d,
                                       int N, int H, int Cc) {
  int node = (int)((blockIdx.x * (size_t)blockDim.x + threadIdx.x) >> 5);
  int lane = threadIdx.x & 31;
  if (node >= N) return;
  int F = H * Cc;
  for (int hd = 0; hd < H; ++hd) {
    float ps = 0.0f, pd = 0.0f;
    for (int c = lane; c < Cc; c += 32) {
      float x = h[(size_t)node * F + hd * Cc + c];
      ps += x * att_src[hd * Cc + c];
      pd += x * att_dst[hd * Cc + c];
    }
    ps = waveReduceSum(ps);
    pd = waveReduceSum(pd);
    if (lane == 0) { a_s[node * H + hd] = ps; a_d[node * H + hd] = pd; }
  }
}

__global__ void fill_kernel(float* __restrict__ p, float v, size_t n) {
  size_t i = blockIdx.x * (size_t)blockDim.x + threadIdx.x;
  if (i < n) p[i] = v;
}

// cew layer: alpha = leaky(s[u]+s[v],0.01)*(cew[u]+cew[v]); segment max.
__global__ void cew_alpha_max_kernel(const int* __restrict__ src,
                                     const int* __restrict__ dst,
                                     const float* __restrict__ s,
                                     const float* __restrict__ cew,
                                     float* __restrict__ alpha,
                                     float* __restrict__ m, int E) {
  int e = blockIdx.x * blockDim.x + threadIdx.x;
  if (e >= E) return;
  int u = src[e], v = dst[e];
  float a = s[u] + s[v];
  a = (a > 0.0f) ? a : 0.01f * a;
  a *= (cew[u] + cew[v]);
  alpha[e] = a;
  atomicMaxF(&m[v], a);
}

// GAT edge scores with self loops appended at idx in [E, Etot).
__global__ void gat_alpha_max_kernel(const int* __restrict__ src,
                                     const int* __restrict__ dst,
                                     const float* __restrict__ a_s,
                                     const float* __restrict__ a_d,
                                     float* __restrict__ alpha,
                                     float* __restrict__ m,
                                     int E, int Etot, int H, float slope) {
  int idx = blockIdx.x * blockDim.x + threadIdx.x;
  if (idx >= Etot) return;
  int u = (idx < E) ? src[idx] : (idx - E);
  int v = (idx < E) ? dst[idx] : (idx - E);
  for (int hd = 0; hd < H; ++hd) {
    float a = a_s[u * H + hd] + a_d[v * H + hd];
    a = (a > 0.0f) ? a : slope * a;
    alpha[(size_t)idx * H + hd] = a;
    atomicMaxF(&m[v * H + hd], a);
  }
}

// Transformer edge scores: alpha[e,h] = dot(q[v,h,:], k[u,h,:]) * scale.
__global__ void tc_alpha_max_kernel(const int* __restrict__ src,
                                    const int* __restrict__ dst,
                                    const float* __restrict__ q,
                                    const float* __restrict__ k,
                                    float* __restrict__ alpha,
                                    float* __restrict__ m,
                                    int E, int H, int Cc, float scale) {
  int e = (int)((blockIdx.x * (size_t)blockDim.x + threadIdx.x) >> 5);
  int lane = threadIdx.x & 31;
  if (e >= E) return;
  int u = src[e], v = dst[e];
  int F = H * Cc;
  for (int hd = 0; hd < H; ++hd) {
    float p = 0.0f;
    for (int c = lane; c < Cc; c += 32)
      p += q[(size_t)v * F + hd * Cc + c] * k[(size_t)u * F + hd * Cc + c];
    p = waveReduceSum(p);
    if (lane == 0) {
      p *= scale;
      alpha[(size_t)e * H + hd] = p;
      atomicMaxF(&m[v * H + hd], p);
    }
  }
}

// pass 2: alpha <- exp(alpha - m_fixed[dst]); dsum[dst] += alpha
__global__ void exp_sum_kernel(const int* __restrict__ dst,
                               float* __restrict__ alpha,
                               const float* __restrict__ m,
                               float* __restrict__ dsum,
                               int E, int Etot, int H) {
  int idx = blockIdx.x * blockDim.x + threadIdx.x;
  if (idx >= Etot) return;
  int v = (idx < E) ? dst[idx] : (idx - E);
  for (int hd = 0; hd < H; ++hd) {
    float mm = m[v * H + hd];
    if (mm < -3.0e38f) mm = 0.0f;   // isfinite fix: only -inf can occur
    float ex = __expf(alpha[(size_t)idx * H + hd] - mm);
    alpha[(size_t)idx * H + hd] = ex;
    atomicAdd(&dsum[v * H + hd], ex);
  }
}

// pass 3: agg[v,:] += h[u,:] * alpha/(dsum+eps).  Wave per edge, lanes over F.
__global__ void scatter_agg_kernel(const int* __restrict__ src,
                                   const int* __restrict__ dst,
                                   const float* __restrict__ alpha,
                                   const float* __restrict__ dsum,
                                   const float* __restrict__ h,
                                   float* __restrict__ agg,
                                   int E, int Etot, int H, int Cc) {
  int e = (int)((blockIdx.x * (size_t)blockDim.x + threadIdx.x) >> 5);
  int lane = threadIdx.x & 31;
  if (e >= Etot) return;
  int u = (e < E) ? src[e] : (e - E);
  int v = (e < E) ? dst[e] : (e - E);
  int F = H * Cc;
  float w[HEADS];
  for (int hd = 0; hd < H; ++hd)
    w[hd] = alpha[(size_t)e * H + hd] / (dsum[v * H + hd] + 1e-16f);
  for (int f = lane; f < F; f += 32)
    atomicAdd(&agg[(size_t)v * F + f], h[(size_t)u * F + f] * w[f / Cc]);
}

// elementwise (+bias[ f%F ]) (+relu), in place
__global__ void bias_relu_kernel(float* __restrict__ p,
                                 const float* __restrict__ bias,
                                 size_t n, int F, int relu) {
  size_t i = blockIdx.x * (size_t)blockDim.x + threadIdx.x;
  if (i >= n) return;
  float o = p[i];
  if (bias) o += bias[i % F];
  if (relu) o = fmaxf(o, 0.0f);
  p[i] = o;
}

// transformer epilogue: out[n,c] = relu(mean_h agg[n,h,c] + skip[n,c])
__global__ void tc_combine_kernel(const float* __restrict__ agg,
                                  const float* __restrict__ skip,
                                  float* __restrict__ out,
                                  int N, int H, int Cc) {
  int i = blockIdx.x * blockDim.x + threadIdx.x;
  if (i >= N * Cc) return;
  int n = i / Cc, c = i - n * Cc;
  float s = 0.0f;
  for (int hd = 0; hd < H; ++hd) s += agg[(size_t)n * H * Cc + hd * Cc + c];
  out[i] = fmaxf(s * (1.0f / H) + skip[i], 0.0f);
}

// ---------------------------------------------------------------------------

static inline int cdiv(long long a, long long b) { return (int)((a + b - 1) / b); }

extern "C" void kernel_launch(void* const* d_in, const int* in_sizes, int n_in,
                              void* d_out, int out_size, void* d_ws, size_t ws_size,
                              hipStream_t stream) {
  (void)in_sizes; (void)n_in; (void)out_size; (void)ws_size;
  const float* x       = (const float*)d_in[0];
  const int*   ei      = (const int*)  d_in[1];
  const float* cew     = (const float*)d_in[2];
  const float* cew_W   = (const float*)d_in[3];
  const float* cew_b   = (const float*)d_in[4];
  const float* cew_att = (const float*)d_in[5];
  const float* W1      = (const float*)d_in[6];
  const float* att_s1  = (const float*)d_in[7];
  const float* att_d1  = (const float*)d_in[8];
  const float* b1      = (const float*)d_in[9];
  const float* W2      = (const float*)d_in[10];
  const float* att_s2  = (const float*)d_in[11];
  const float* att_d2  = (const float*)d_in[12];
  const float* b2      = (const float*)d_in[13];
  const float* Wq      = (const float*)d_in[14];
  const float* bq      = (const float*)d_in[15];
  const float* Wk      = (const float*)d_in[16];
  const float* bk      = (const float*)d_in[17];
  const float* Wv      = (const float*)d_in[18];
  const float* bv      = (const float*)d_in[19];
  const float* Wskip   = (const float*)d_in[20];
  const float* bskip   = (const float*)d_in[21];
  const float* Wf      = (const float*)d_in[22];
  const float* bf      = (const float*)d_in[23];
  const int* src = ei;
  const int* dst = ei + Ee;

  // ---- workspace layout (floats) : ~241 MB total -------------------------
  float* ws     = (float*)d_ws;
  const size_t BIG = (size_t)Nn * 256;
  float* bufH   = ws;               // [N,256] transformed features h
  float* bufAgg = bufH   + BIG;     // [N,256] aggregation / g ping
  float* bufX   = bufAgg + BIG;     // [N,256] g pong
  float* bufV   = bufX   + BIG;     // [N,256] v (transformer)
  float* sbuf   = bufV   + BIG;     // [N]
  float* a_s    = sbuf   + Nn;      // [N,4]
  float* a_d    = a_s    + (size_t)Nn * 4;   // [N,4]
  float* mmax   = a_d    + (size_t)Nn * 4;   // [N,4]
  float* dsum   = mmax   + (size_t)Nn * 4;   // [N,4]
  float* alpha  = dsum   + (size_t)Nn * 4;   // [(E+N),4]
  float* skipb  = alpha  + (size_t)(Ee + Nn) * 4; // [N,64]
  float* tbuf   = skipb  + (size_t)Nn * 64;  // [N,64]
  // packed-weight scratch (bf16), 32B aligned: all prior offsets are
  // multiples of 8 floats.  Max weight = 128*256 = 32K bf16 = 64 KB.
  __bf16* packB = (__bf16*)(tbuf + (size_t)Nn * 64);

  const float NEG_INF = -__builtin_inff();
  const int E2 = Ee + Nn;
  const int TB = 256;                       // generic block (8 waves)
  const int GEMM_TB = 128;                  // 4 waves/block

  // GEMM dispatch: pack W -> bf16 B-fragment order, then 16x64 strips (CT=4)
  // when Nout % 64 == 0, else 16x32 strips (CT=2, final Nout=32 GEMM).
  auto gemm = [&](const float* A, const float* Wm, const float* bias, float* C,
                  int M, int K, int Nout) {
    int packTotal = (Nout / 16) * (K / 32) * 32;
    pack_b_kernel<<<cdiv(packTotal, TB), TB, 0, stream>>>(Wm, packB, K, Nout);
    if (Nout % 64 == 0) {
      int strips = (M / 16) * (Nout / 64);
      int blocks = cdiv(strips, GEMM_TB / 32);
      if (bias)
        gemm_wmma_kernel<4, true ><<<blocks, GEMM_TB, 0, stream>>>(A, packB, bias, C, M, K, Nout);
      else
        gemm_wmma_kernel<4, false><<<blocks, GEMM_TB, 0, stream>>>(A, packB, bias, C, M, K, Nout);
    } else {
      int strips = (M / 16) * (Nout / 32);
      int blocks = cdiv(strips, GEMM_TB / 32);
      if (bias)
        gemm_wmma_kernel<2, true ><<<blocks, GEMM_TB, 0, stream>>>(A, packB, bias, C, M, K, Nout);
      else
        gemm_wmma_kernel<2, false><<<blocks, GEMM_TB, 0, stream>>>(A, packB, bias, C, M, K, Nout);
    }
  };

  // ============== Layer 0: cew_gat (F=128, H=1) ==========================
  gemm(x, cew_W, cew_b, bufH, Nn, DIN, H0);                          // h0
  matvec128_kernel<<<cdiv((long long)Nn * 32, TB), TB, 0, stream>>>(bufH, cew_att, sbuf, Nn);
  fill_kernel<<<cdiv(Nn, TB), TB, 0, stream>>>(mmax, NEG_INF, (size_t)Nn);
  fill_kernel<<<cdiv(Nn, TB), TB, 0, stream>>>(dsum, 0.0f, (size_t)Nn);
  fill_kernel<<<cdiv((long long)Nn * H0, TB), TB, 0, stream>>>(bufAgg, 0.0f, (size_t)Nn * H0);
  cew_alpha_max_kernel<<<cdiv(Ee, TB), TB, 0, stream>>>(src, dst, sbuf, cew, alpha, mmax, Ee);
  exp_sum_kernel<<<cdiv(Ee, TB), TB, 0, stream>>>(dst, alpha, mmax, dsum, Ee, Ee, 1);
  scatter_agg_kernel<<<cdiv((long long)Ee * 32, TB), TB, 0, stream>>>(
      src, dst, alpha, dsum, bufH, bufAgg, Ee, Ee, 1, H0);
  bias_relu_kernel<<<cdiv((long long)Nn * H0, TB), TB, 0, stream>>>(
      bufAgg, nullptr, (size_t)Nn * H0, H0, 1);                      // g0 in bufAgg

  // ============== Layer 1: GAT (H=4, C=64 -> F=256) ======================
  gemm(bufAgg, W1, nullptr, bufH, Nn, H0, HEADS * C1);               // h1
  gat_node_scores_kernel<<<cdiv((long long)Nn * 32, TB), TB, 0, stream>>>(
      bufH, att_s1, att_d1, a_s, a_d, Nn, HEADS, C1);
  fill_kernel<<<cdiv(Nn * 4, TB), TB, 0, stream>>>(mmax, NEG_INF, (size_t)Nn * 4);
  fill_kernel<<<cdiv(Nn * 4, TB), TB, 0, stream>>>(dsum, 0.0f, (size_t)Nn * 4);
  fill_kernel<<<cdiv((long long)Nn * 256, TB), TB, 0, stream>>>(bufX, 0.0f, (size_t)Nn * 256);
  gat_alpha_max_kernel<<<cdiv(E2, TB), TB, 0, stream>>>(
      src, dst, a_s, a_d, alpha, mmax, Ee, E2, HEADS, 0.2f);
  exp_sum_kernel<<<cdiv(E2, TB), TB, 0, stream>>>(dst, alpha, mmax, dsum, Ee, E2, HEADS);
  scatter_agg_kernel<<<cdiv((long long)E2 * 32, TB), TB, 0, stream>>>(
      src, dst, alpha, dsum, bufH, bufX, Ee, E2, HEADS, C1);
  bias_relu_kernel<<<cdiv((long long)Nn * 256, TB), TB, 0, stream>>>(
      bufX, b1, (size_t)Nn * 256, 256, 1);                           // g1 in bufX

  // ============== Layer 2: GAT (H=4, C=32 -> F=128) ======================
  gemm(bufX, W2, nullptr, bufH, Nn, HEADS * C1, HEADS * C2);         // h2
  gat_node_scores_kernel<<<cdiv((long long)Nn * 32, TB), TB, 0, stream>>>(
      bufH, att_s2, att_d2, a_s, a_d, Nn, HEADS, C2);
  fill_kernel<<<cdiv(Nn * 4, TB), TB, 0, stream>>>(mmax, NEG_INF, (size_t)Nn * 4);
  fill_kernel<<<cdiv(Nn * 4, TB), TB, 0, stream>>>(dsum, 0.0f, (size_t)Nn * 4);
  fill_kernel<<<cdiv((long long)Nn * 128, TB), TB, 0, stream>>>(bufAgg, 0.0f, (size_t)Nn * 128);
  gat_alpha_max_kernel<<<cdiv(E2, TB), TB, 0, stream>>>(
      src, dst, a_s, a_d, alpha, mmax, Ee, E2, HEADS, 0.2f);
  exp_sum_kernel<<<cdiv(E2, TB), TB, 0, stream>>>(dst, alpha, mmax, dsum, Ee, E2, HEADS);
  scatter_agg_kernel<<<cdiv((long long)E2 * 32, TB), TB, 0, stream>>>(
      src, dst, alpha, dsum, bufH, bufAgg, Ee, E2, HEADS, C2);
  bias_relu_kernel<<<cdiv((long long)Nn * 128, TB), TB, 0, stream>>>(
      bufAgg, b2, (size_t)Nn * 128, 128, 1);                         // g2 in bufAgg

  // ============== Layer 3: TransformerConv (H=4, C=64) ===================
  gemm(bufAgg, Wq, bq, bufH, Nn, 128, HEADS * C3);                   // q
  gemm(bufAgg, Wk, bk, bufX, Nn, 128, HEADS * C3);                   // k
  gemm(bufAgg, Wv, bv, bufV, Nn, 128, HEADS * C3);                   // v
  gemm(bufAgg, Wskip, bskip, skipb, Nn, 128, C3);                    // skip (g2 dead after)
  fill_kernel<<<cdiv(Nn * 4, TB), TB, 0, stream>>>(mmax, NEG_INF, (size_t)Nn * 4);
  fill_kernel<<<cdiv(Nn * 4, TB), TB, 0, stream>>>(dsum, 0.0f, (size_t)Nn * 4);
  tc_alpha_max_kernel<<<cdiv((long long)Ee * 32, TB), TB, 0, stream>>>(
      src, dst, bufH, bufX, alpha, mmax, Ee, HEADS, C3, 0.125f);     // 1/sqrt(64)
  exp_sum_kernel<<<cdiv(Ee, TB), TB, 0, stream>>>(dst, alpha, mmax, dsum, Ee, Ee, HEADS);
  fill_kernel<<<cdiv((long long)Nn * 256, TB), TB, 0, stream>>>(bufAgg, 0.0f, (size_t)Nn * 256);
  scatter_agg_kernel<<<cdiv((long long)Ee * 32, TB), TB, 0, stream>>>(
      src, dst, alpha, dsum, bufV, bufAgg, Ee, Ee, HEADS, C3);
  tc_combine_kernel<<<cdiv(Nn * C3, TB), TB, 0, stream>>>(bufAgg, skipb, tbuf, Nn, HEADS, C3);

  // ============== Final linear: out = t @ Wf + bf ========================
  gemm(tbuf, Wf, bf, (float*)d_out, Nn, C3, DOUT);
}
